// NLSTM_20057497272914
// MI455X (gfx1250) — compile-verified, matrix operand order checked by reading, not run on hardware
//
#include <hip/hip_runtime.h>
#include <hip/hip_bf16.h>

typedef __bf16 bf16;
typedef __attribute__((ext_vector_type(16))) __bf16 v16bf;
typedef __attribute__((ext_vector_type(8)))  float  v8f;

#define B_   64
#define T_   48
#define N_   325
#define F_   128
#define H_   128
#define BN_  (B_ * N_)       // 20800
#define G4_  (4 * H_)        // 512

__device__ __forceinline__ float sigf(float x) { return 1.0f / (1.0f + __expf(-x)); }
__device__ __forceinline__ float tanhfast(float x) { return 2.0f * sigf(2.0f * x) - 1.0f; }

// Convert layer-1 weights to bf16, fold the two biases together.
__global__ void nlstm_prep(const float* __restrict__ W_ih, const float* __restrict__ W_hh,
                           const float* __restrict__ b_ih, const float* __restrict__ b_hh,
                           bf16* __restrict__ wih_bf, bf16* __restrict__ whh_bf,
                           float* __restrict__ bias) {
    int i = blockIdx.x * blockDim.x + threadIdx.x;
    const int WOFF = G4_ * F_;   // layer index 1
    if (i < G4_ * F_) {
        wih_bf[i] = (bf16)W_ih[WOFF + i];
        whh_bf[i] = (bf16)W_hh[WOFF + i];
    }
    if (i < G4_) bias[i] = b_ih[G4_ + i] + b_hh[G4_ + i];
}

__global__ void nlstm_zero(bf16* __restrict__ h, float* __restrict__ c, int n) {
    int i = blockIdx.x * blockDim.x + threadIdx.x;
    if (i < n) { h[i] = (bf16)0.0f; c[i] = 0.0f; }
}

// One LSTM timestep for layer 1. Block = 8 waves = 16 rows x all 512 gate cols.
__global__ __launch_bounds__(256) void nlstm_step(
    const float* __restrict__ x,      // [B, T, N, F] fp32 (original layout)
    const bf16*  __restrict__ wih,    // [512, 128] bf16
    const bf16*  __restrict__ whh,    // [512, 128] bf16
    const float* __restrict__ bias,   // [512]
    bf16*  __restrict__ hbuf,         // [BN, 128] bf16 (recurrent input/output)
    float* __restrict__ cbuf,         // [BN, 128] fp32
    float* __restrict__ out,          // [BN, 128] fp32
    int t, int last)
{
    const int wave  = threadIdx.x >> 5;
    const int lane  = threadIdx.x & 31;
    const int r0    = blockIdx.x * 16;      // row-tile base in BN
    const int j0    = wave * 16;            // this wave's 16 H-columns

    // A-fragment addressing: lanes 0-15 -> rows 0-15 / K 0-15 of the 32-K slab,
    // lanes 16-31 -> same rows / K 16-31. B mirrors it with lane%16 = column.
    const int  mrow  = lane & 15;
    const int  khalf = (lane >> 4) * 16;
    const int  r     = r0 + mrow;
    const int  b     = r / N_;
    const int  n     = r - b * N_;
    const long xbase = ((long)(b * T_ + t) * N_ + n) * F_;   // x[b, t, n, :]

    v8f acc[4] = {};   // i, f, g, o accumulator tiles (identical layouts)

    #pragma unroll
    for (int kk = 0; kk < 4; ++kk) {       // K = 128 = 4 x 32
        const int kb = kk * 32 + khalf;

        // A from x: 16 contiguous fp32 -> bf16
        v16bf ax;
        const float* xp = x + xbase + kb;
        #pragma unroll
        for (int e = 0; e < 16; ++e) ax[e] = (bf16)xp[e];

        // A from h(t-1): 16 contiguous bf16 (32 bytes)
        v16bf ah = *(const v16bf*)(hbuf + (long)r * H_ + kb);

        #pragma unroll
        for (int gi = 0; gi < 4; ++gi) {
            const int col = gi * H_ + j0 + mrow;     // gate-column (row of W)
            v16bf bi = *(const v16bf*)(wih + (long)col * F_ + kb);
            v16bf bh = *(const v16bf*)(whh + (long)col * H_ + kb);
            acc[gi] = __builtin_amdgcn_wmma_f32_16x16x32_bf16(
                false, ax, false, bi, (short)0, acc[gi], false, false);
            acc[gi] = __builtin_amdgcn_wmma_f32_16x16x32_bf16(
                false, ah, false, bh, (short)0, acc[gi], false, false);
        }
    }

    // All waves have finished reading h(t-1) rows r0..r0+15 before any write.
    __syncthreads();

    // C/D layout: vgpr v, lane l -> row = v + 8*(l/16), col = l%16.
    const int ncol = lane & 15;
    const int mb   = (lane >> 4) * 8;
    const float bI = bias[0 * H_ + j0 + ncol];
    const float bF = bias[1 * H_ + j0 + ncol];
    const float bG = bias[2 * H_ + j0 + ncol];
    const float bO = bias[3 * H_ + j0 + ncol];

    #pragma unroll
    for (int v = 0; v < 8; ++v) {
        const int  rr  = r0 + mb + v;
        const long idx = (long)rr * H_ + j0 + ncol;
        const float ig = sigf(acc[0][v] + bI);
        const float fg = sigf(acc[1][v] + bF);
        const float gg = tanhfast(acc[2][v] + bG);
        const float og = sigf(acc[3][v] + bO);
        const float cn = fg * cbuf[idx] + ig * gg;
        const float hn = og * tanhfast(cn);
        cbuf[idx] = cn;
        hbuf[idx] = (bf16)hn;
        if (last) out[idx] = hn;
    }
}

extern "C" void kernel_launch(void* const* d_in, const int* in_sizes, int n_in,
                              void* d_out, int out_size, void* d_ws, size_t ws_size,
                              hipStream_t stream) {
    const float* x    = (const float*)d_in[0];
    const float* W_ih = (const float*)d_in[1];
    const float* W_hh = (const float*)d_in[2];
    const float* b_ih = (const float*)d_in[3];
    const float* b_hh = (const float*)d_in[4];
    float* out = (float*)d_out;

    char* ws = (char*)d_ws;
    // Workspace layout (bytes):
    //   [0, 128K)        Wih bf16   512*128*2
    //   [128K, 256K)     Whh bf16
    //   [256K, 258K)     bias fp32  512*4
    //   [264192, +5.3MB) h bf16     20800*128*2
    //   [..., +10.6MB)   c fp32     20800*128*4
    bf16*  wih_bf = (bf16*) (ws);
    bf16*  whh_bf = (bf16*) (ws + 131072);
    float* bias   = (float*)(ws + 262144);
    bf16*  hbuf   = (bf16*) (ws + 264192);
    float* cbuf   = (float*)(ws + 264192 + (size_t)BN_ * H_ * 2);

    nlstm_prep<<<(G4_ * F_ + 255) / 256, 256, 0, stream>>>(
        W_ih, W_hh, b_ih, b_hh, wih_bf, whh_bf, bias);
    nlstm_zero<<<(BN_ * H_ + 255) / 256, 256, 0, stream>>>(hbuf, cbuf, BN_ * H_);

    for (int t = 0; t < T_; ++t) {
        nlstm_step<<<BN_ / 16, 256, 0, stream>>>(
            x, wih_bf, whh_bf, bias, hbuf, cbuf, out, t, (t == T_ - 1) ? 1 : 0);
    }
}
